// HSIteration_ApriPolyEncDec_23510650978742
// MI455X (gfx1250) — matrix operand order repaired and checked by reading, not compiled, not run on hardware
//
#include <hip/hip_runtime.h>
#include <hip/hip_bf16.h>
#include <utility>

// ---------------------------------------------------------------------------
// MI455X / gfx1250 implementation. All dense N x 64 x 64 contractions run on
// v_wmma_f32_16x16x32_f16 (wave32).  fp32 master copies + fp32 accumulation;
// global Frobenius scalars folded into the tiny 64x64 matrices so f16
// operands stay well-scaled.
// ---------------------------------------------------------------------------

typedef __attribute__((ext_vector_type(16))) _Float16 v16h;
typedef __attribute__((ext_vector_type(8)))  _Float16 v8h;
typedef __attribute__((ext_vector_type(8)))  float    v8f;

#define DDIM 64
#define RCHUNK 1024

// scalar slots in workspace
#define SC_SSQQ   0
#define SC_SSQK   1
#define SC_FRQ    2
#define SC_FRK    3
#define SC_INVFQK 4
#define SC_T1     5
#define SC_T2     6
#define SC_T3     7
#define SC_ONE    8
#define SC_KSUM   16   // 64 floats
#define SC_CSUM   80   // 64 floats
#define SC_SCH    144  // 64 floats (per-chunk Frobenius norms)

static __device__ __forceinline__ v8f wmma16x16x32(v16h a, v16h b, v8f c) {
  return __builtin_amdgcn_wmma_f32_16x16x32_f16(false, a, false, b, (short)0, c,
                                                false, false);
}

static __device__ __forceinline__ float wave_reduce(float v) {
#pragma unroll
  for (int m = 16; m > 0; m >>= 1) v += __shfl_xor(v, m, 32);
  return v;
}

// load 16 halves (converted from fp32) from two 8-contiguous segments
static __device__ __forceinline__ v16h cvt16(const float* p0, const float* p1) {
  v16h r;
#pragma unroll
  for (int i = 0; i < 8; ++i) {
    r[i]     = (_Float16)p0[i];
    r[i + 8] = (_Float16)p1[i];
  }
  return r;
}

// ---------------------------------------------------------------------------
__global__ void k_zero(float* __restrict__ p, int n) {
  int i = blockIdx.x * blockDim.x + threadIdx.x;
  if (i < n) p[i] = 0.f;
}

// dstH = f16(src); dst32 = src; acc (+)= coef*src
__global__ void k_fuse(const float* __restrict__ src, float* __restrict__ dst32,
                       _Float16* __restrict__ dstH, float* __restrict__ acc,
                       const float* __restrict__ coef, int accWrite, int n) {
  int i = blockIdx.x * blockDim.x + threadIdx.x;
  if (i >= n) return;
  float v = src[i];
  if (dst32) dst32[i] = v;
  if (dstH) dstH[i] = (_Float16)v;
  if (acc) {
    float c = coef[0];
    if (accWrite) acc[i] = c * v;
    else          acc[i] += c * v;
  }
}

// ---------------------------------------------------------------------------
// Fused Q/K projection: Q = preds @ Wq^T + bq ; K likewise.  WMMA per tile.
// Also accumulates sumsq(Q), sumsq(K), colsum(K) (fp32).
__global__ __launch_bounds__(256) void k_proj(
    const float* __restrict__ preds, const float* __restrict__ Wq,
    const float* __restrict__ bq, const float* __restrict__ Wk,
    const float* __restrict__ bk, float* __restrict__ Q32,
    _Float16* __restrict__ QH, _Float16* __restrict__ KH,
    float* __restrict__ SC) {
  __shared__ float lds_ks[DDIM];
  __shared__ float lds_ssq[2];
  int tid = threadIdx.x;
  if (tid < DDIM) lds_ks[tid] = 0.f;
  if (tid < 2) lds_ssq[tid] = 0.f;
  __syncthreads();

  int wave = tid >> 5, lane = tid & 31;
  int laneHi = lane >> 4, ln = lane & 15;
  int rowHalf = wave >> 2, nTile = wave & 3;
  int nodeBase = blockIdx.x * 32 + rowHalf * 16;
  int node = nodeBase + ln;       // A row (node)
  int dout = nTile * 16 + ln;     // B col (output dim)

  v8f cq = {}; v8f ck = {};
#pragma unroll
  for (int kc = 0; kc < DDIM; kc += 32) {
    int aoff = kc + laneHi * 16;
    v16h a = cvt16(&preds[(size_t)node * DDIM + aoff],
                   &preds[(size_t)node * DDIM + aoff + 8]);
    const float* wq = &Wq[(size_t)dout * DDIM + kc + laneHi * 16];
    const float* wk = &Wk[(size_t)dout * DDIM + kc + laneHi * 16];
    v16h bqv = cvt16(wq, wq + 8);
    v16h bkv = cvt16(wk, wk + 8);
    cq = wmma16x16x32(a, bqv, cq);
    ck = wmma16x16x32(a, bkv, ck);
  }

  float biasq = bq[dout], biask = bk[dout];
  float ssq_q = 0.f, ssq_k = 0.f, kcol = 0.f;
#pragma unroll
  for (int r = 0; r < 8; ++r) {
    int nrow = nodeBase + r + 8 * laneHi;  // C layout: M = r + 8*laneHi
    float qv = cq[r] + biasq;
    float kv = ck[r] + biask;
    Q32[(size_t)nrow * DDIM + dout] = qv;
    QH[(size_t)nrow * DDIM + dout] = (_Float16)qv;
    KH[(size_t)nrow * DDIM + dout] = (_Float16)kv;
    ssq_q += qv * qv;
    ssq_k += kv * kv;
    kcol += kv;
  }
  atomicAdd(&lds_ks[dout], kcol);
  ssq_q = wave_reduce(ssq_q);
  ssq_k = wave_reduce(ssq_k);
  if (lane == 0) {
    atomicAdd(&lds_ssq[0], ssq_q);
    atomicAdd(&lds_ssq[1], ssq_k);
  }
  __syncthreads();
  if (tid < DDIM) atomicAdd(&SC[SC_KSUM + tid], lds_ks[tid]);
  if (tid == 64) atomicAdd(&SC[SC_SSQQ], lds_ssq[0]);
  if (tid == 65) atomicAdd(&SC[SC_SSQK], lds_ssq[1]);
}

__global__ void k_finalize(float* SC) {
  float frq = sqrtf(SC[SC_SSQQ]);
  float frk = sqrtf(SC[SC_SSQK]);
  SC[SC_FRQ] = frq;
  SC[SC_FRK] = frk;
  SC[SC_INVFQK] = 1.f / (frq * frk);
  SC[SC_ONE] = 1.f;
}

// out[row] = dot(src[row,:], vec) * (*scalePtr) + addConst   (one wave / row)
__global__ __launch_bounds__(256) void k_rowdot(
    const float* __restrict__ src, const float* __restrict__ vec,
    const float* __restrict__ scalePtr, float addConst, float* __restrict__ out,
    int N) {
  int row = (blockIdx.x * 256 + threadIdx.x) >> 5;
  int lane = threadIdx.x & 31;
  if (row >= N) return;
  float p = src[(size_t)row * DDIM + lane] * vec[lane] +
            src[(size_t)row * DDIM + lane + 32] * vec[lane + 32];
  p = wave_reduce(p);
  if (lane == 0) out[row] = p * scalePtr[0] + addConst;
}

// ---------------------------------------------------------------------------
// OUT(64x64) += X^T Y over N nodes (WMMA reduction, LDS-transposed staging).
// Optional colsum += column sums of Ysrc (fp32 master copy).
#define ATB_NODES 1024
__global__ __launch_bounds__(256) void k_atb(
    const _Float16* __restrict__ X, const _Float16* __restrict__ Y,
    const float* __restrict__ Ysrc, float* __restrict__ OUT,
    float* __restrict__ colsum) {
  __shared__ _Float16 Xt[DDIM * 32];  // [dim][node]
  __shared__ _Float16 Yt[DDIM * 32];
  __shared__ float lds_cs[DDIM];
  int tid = threadIdx.x, wave = tid >> 5, lane = tid & 31;
  int laneHi = lane >> 4, ln = lane & 15;
  size_t nodeBase = (size_t)blockIdx.x * ATB_NODES;

  if (Ysrc) {
    if (tid < DDIM) lds_cs[tid] = 0.f;
    __syncthreads();
    int dim = tid & 63;
    float s = 0.f;
    for (int n = tid >> 6; n < ATB_NODES; n += 4)
      s += Ysrc[(nodeBase + n) * DDIM + dim];
    atomicAdd(&lds_cs[dim], s);
    __syncthreads();
    if (tid < DDIM) atomicAdd(&colsum[tid], lds_cs[tid]);
  }

  int t0 = 2 * wave;               // two 16x16 output tiles per wave
  int mTile = t0 >> 2, n0 = t0 & 3, n1 = n0 + 1;
  v8f c0 = {}; v8f c1 = {};

  for (int kc = 0; kc < ATB_NODES; kc += 32) {
    __syncthreads();
    {  // stage 32 nodes x 64 dims, transposed into LDS
      int idx = tid * 8;
      int nd = idx >> 6, dm = idx & 63;
      const v8h xv = *(const v8h*)&X[(nodeBase + kc + nd) * DDIM + dm];
      const v8h yv = *(const v8h*)&Y[(nodeBase + kc + nd) * DDIM + dm];
#pragma unroll
      for (int q = 0; q < 8; ++q) {
        Xt[(dm + q) * 32 + nd] = xv[q];
        Yt[(dm + q) * 32 + nd] = yv[q];
      }
    }
    __syncthreads();
    int adim = mTile * 16 + ln;
    union { v16h v; v8h h[2]; } ua;
    ua.h[0] = *(const v8h*)&Xt[adim * 32 + laneHi * 16];
    ua.h[1] = *(const v8h*)&Xt[adim * 32 + laneHi * 16 + 8];
    v16h b0 = *(const v16h*)&Yt[(n0 * 16 + ln) * 32 + laneHi * 16];
    v16h b1 = *(const v16h*)&Yt[(n1 * 16 + ln) * 32 + laneHi * 16];
    c0 = wmma16x16x32(ua.v, b0, c0);
    c1 = wmma16x16x32(ua.v, b1, c1);
  }
#pragma unroll
  for (int r = 0; r < 8; ++r) {
    int m = mTile * 16 + r + 8 * laneHi;
    atomicAdd(&OUT[m * DDIM + n0 * 16 + ln], c0[r]);
    atomicAdd(&OUT[m * DDIM + n1 * 16 + ln], c1[r]);
  }
}

// B2H = f16( (C_raw @ KV) / (frq*frk) )   (tiny 64^3)
__global__ void k_stepB(const float* __restrict__ C, const float* __restrict__ KV,
                        const float* __restrict__ SC, _Float16* __restrict__ B2H) {
  int o = blockIdx.x * blockDim.x + threadIdx.x;
  if (o >= DDIM * DDIM) return;
  int m = o >> 6, d = o & 63;
  float s = 0.f;
#pragma unroll 8
  for (int j = 0; j < DDIM; ++j) s += C[m * DDIM + j] * KV[j * DDIM + d];
  B2H[o] = (_Float16)(s * SC[SC_INVFQK]);
}

// OUT[n,:] = (Qh @ B2H * invfqk + csum) / z[n]; optional f16 copy + ACC += b*OUT
__global__ __launch_bounds__(256) void k_bcast(
    const _Float16* __restrict__ QH, const _Float16* __restrict__ B2H,
    const float* __restrict__ SC, const float* __restrict__ Z,
    float* __restrict__ OUT32, _Float16* __restrict__ OUTH,
    float* __restrict__ ACC, const float* __restrict__ coef) {
  __shared__ _Float16 Bt[DDIM * DDIM];  // transposed: Bt[d][m]
  int tid = threadIdx.x;
  for (int idx = tid; idx < DDIM * DDIM; idx += 256) {
    int m = idx >> 6, d = idx & 63;
    Bt[d * DDIM + m] = B2H[idx];
  }
  __syncthreads();

  int wave = tid >> 5, lane = tid & 31, laneHi = lane >> 4, ln = lane & 15;
  int rowHalf = wave >> 2, nTile = wave & 3;
  int nodeBase = blockIdx.x * 32 + rowHalf * 16;
  int node = nodeBase + ln;
  int dim = nTile * 16 + ln;

  v8f c = {};
#pragma unroll
  for (int kc = 0; kc < DDIM; kc += 32) {
    union { v16h v; v8h h[2]; } ua;
    ua.h[0] = *(const v8h*)&QH[(size_t)node * DDIM + kc + laneHi * 16];
    ua.h[1] = *(const v8h*)&QH[(size_t)node * DDIM + kc + laneHi * 16 + 8];
    v16h b = *(const v16h*)&Bt[dim * DDIM + kc + laneHi * 16];
    c = wmma16x16x32(ua.v, b, c);
  }
  float inv = SC[SC_INVFQK];
  float cs = SC[SC_CSUM + dim];
  float beta = coef ? coef[0] : 0.f;
#pragma unroll
  for (int r = 0; r < 8; ++r) {
    int nrow = nodeBase + r + 8 * laneHi;
    float v = (c[r] * inv + cs) / Z[nrow];
    OUT32[(size_t)nrow * DDIM + dim] = v;
    if (OUTH) OUTH[(size_t)nrow * DDIM + dim] = (_Float16)v;
    if (ACC) ACC[(size_t)nrow * DDIM + dim] += beta * v;
  }
}

// COO scatter SpMM: OUT[row,:] += val * X[col,:]
__global__ void k_spmm(const int* __restrict__ rows, const int* __restrict__ cols,
                       const float* __restrict__ vals, const float* __restrict__ X,
                       float* __restrict__ OUT, int nnz) {
  int gid = blockIdx.x * 256 + threadIdx.x;
  int e = gid >> 4;
  if (e >= nnz) return;
  if (e + 8192 < nnz) __builtin_prefetch(&rows[e + 8192], 0, 1);
  int part = gid & 15;
  int r = rows[e], c = cols[e];
  float v = vals[e];
  const float4 m = *(const float4*)&X[(size_t)c * DDIM + part * 4];
  float* o = &OUT[(size_t)r * DDIM + part * 4];
  atomicAdd(o + 0, v * m.x);
  atomicAdd(o + 1, v * m.y);
  atomicAdd(o + 2, v * m.z);
  atomicAdd(o + 3, v * m.w);
}

// per-chunk Frobenius norm of gq (RCHUNK rows each)
__global__ __launch_bounds__(256) void k_chunknorm(const float* __restrict__ G,
                                                   float* __restrict__ SCH) {
  __shared__ float red[256];
  const float* p = G + (size_t)blockIdx.x * RCHUNK * DDIM;
  float s = 0.f;
  for (int i = threadIdx.x; i < RCHUNK * DDIM; i += 256) {
    float v = p[i];
    s += v * v;
  }
  red[threadIdx.x] = s;
  __syncthreads();
  for (int st = 128; st > 0; st >>= 1) {
    if (threadIdx.x < st) red[threadIdx.x] += red[threadIdx.x + st];
    __syncthreads();
  }
  if (threadIdx.x == 0) SCH[blockIdx.x] = sqrtf(red[0]);
}

// t1 = sum_n (gq_n G gq_n) / (s^2 frk^2 normz^2)
__global__ __launch_bounds__(256) void k_t1(const float* __restrict__ GQ,
                                            const float* __restrict__ GRAW,
                                            const float* __restrict__ Wn,
                                            float* __restrict__ SC, int N) {
  __shared__ float Gs[DDIM * DDIM];
  __shared__ float rowbuf[8][DDIM];
  __shared__ float t1red;
  for (int i = threadIdx.x; i < DDIM * DDIM; i += 256) Gs[i] = GRAW[i];
  if (threadIdx.x == 0) t1red = 0.f;
  __syncthreads();
  int wave = threadIdx.x >> 5, lane = threadIdx.x & 31;
  float frk = SC[SC_FRK];
  float local = 0.f;
  for (int base = blockIdx.x * 8; base < N; base += gridDim.x * 8) {
    int row = base + wave;
    if (row < N) {
      rowbuf[wave][lane] = GQ[(size_t)row * DDIM + lane];
      rowbuf[wave][lane + 32] = GQ[(size_t)row * DDIM + lane + 32];
    }
    __syncthreads();
    if (row < N) {
      const float* rb = rowbuf[wave];
      float h1 = 0.f, h2 = 0.f;
#pragma unroll 8
      for (int i = 0; i < DDIM; ++i) {
        float g = rb[i];
        h1 += g * Gs[i * DDIM + lane];
        h2 += g * Gs[i * DDIM + lane + 32];
      }
      float p = rb[lane] * h1 + rb[lane + 32] * h2;
      p = wave_reduce(p);
      if (lane == 0) {
        float sf = SC[SC_SCH + (row >> 10)] * frk;
        float nz = Wn[row] / sf + (float)RCHUNK;
        local += p / (sf * sf * nz * nz);
      }
    }
    __syncthreads();
  }
  if (lane == 0) atomicAdd(&t1red, local);
  __syncthreads();
  if (threadIdx.x == 0) atomicAdd(&SC[SC_T1], t1red);
}

// t2 = 2 c0 * sum_e vals * (qn2[r].kn[c]) / normz[r] ;  t3raw = sum vals^2
__global__ __launch_bounds__(256) void k_t2t3(
    const int* __restrict__ rows, const int* __restrict__ cols,
    const float* __restrict__ vals, const float* __restrict__ GQ,
    const _Float16* __restrict__ KH, const float* __restrict__ Wn,
    float* __restrict__ SC, const float* __restrict__ beta, int nnz) {
  __shared__ float red2, red3;
  if (threadIdx.x == 0) { red2 = 0.f; red3 = 0.f; }
  __syncthreads();
  int gid = blockIdx.x * 256 + threadIdx.x;
  int e = gid >> 3, sub = gid & 7;
  if (e < nnz) {
    int r = rows[e], c = cols[e];
    float p = 0.f;
#pragma unroll
    for (int k = 0; k < 8; ++k) {
      int d = sub + 8 * k;
      p += GQ[(size_t)r * DDIM + d] * (float)KH[(size_t)c * DDIM + d];
    }
    p += __shfl_xor(p, 1, 8);
    p += __shfl_xor(p, 2, 8);
    p += __shfl_xor(p, 4, 8);
    if (sub == 0) {
      float frk = SC[SC_FRK];
      float sf = SC[SC_SCH + (r >> 10)] * frk;
      float nz = Wn[r] / sf + (float)RCHUNK;
      float v = vals[e];
      float c0 = beta[0] - 1.f;
      atomicAdd(&red2, 2.f * c0 * v * (p / sf) / nz);
      atomicAdd(&red3, v * v);
    }
  }
  __syncthreads();
  if (threadIdx.x == 0) {
    atomicAdd(&SC[SC_T2], red2);
    atomicAdd(&SC[SC_T3], red3);
  }
}

__global__ void k_final(const float* SC, const float* beta, float* out) {
  float c0 = beta[0] - 1.f;
  out[0] = sqrtf(SC[SC_T1] + SC[SC_T2] + c0 * c0 * SC[SC_T3]);
}

// ---------------------------------------------------------------------------
extern "C" void kernel_launch(void* const* d_in, const int* in_sizes, int n_in,
                              void* d_out, int out_size, void* d_ws,
                              size_t ws_size, hipStream_t stream) {
  const float* preds = (const float*)d_in[0];
  const int* rows = (const int*)d_in[2];
  const int* cols = (const int*)d_in[3];
  const float* vals = (const float*)d_in[4];
  const float* Wq = (const float*)d_in[5];
  const float* bq = (const float*)d_in[6];
  const float* Wk = (const float*)d_in[7];
  const float* bk = (const float*)d_in[8];
  const float* lin1 = (const float*)d_in[9];  // alph[4]
  const float* lin2 = (const float*)d_in[10]; // beta[4]
  float* out = (float*)d_out;

  const int N = in_sizes[0] / DDIM;
  const int nnz = in_sizes[2];
  const int ND = N * DDIM;

  // workspace bump allocator
  char* wptr = (char*)d_ws;
  auto alloc = [&](size_t bytes) {
    void* p = (void*)wptr;
    wptr += (bytes + 255) & ~(size_t)255;
    return p;
  };
  float* SC = (float*)alloc(256 * 4);
  float* C_RAW = (float*)alloc(4096 * 4);
  float* KV = (float*)alloc(4096 * 4);
  float* GRAW = (float*)alloc(4096 * 4);
  _Float16* B2H = (_Float16*)alloc(4096 * 2);
  float* Z = (float*)alloc((size_t)N * 4);
  float* Wn = (float*)alloc((size_t)N * 4);
  float* Q32 = (float*)alloc((size_t)ND * 4);
  _Float16* QH = (_Float16*)alloc((size_t)ND * 2);
  _Float16* KH = (_Float16*)alloc((size_t)ND * 2);
  float* VA32 = (float*)alloc((size_t)ND * 4);
  _Float16* VAH = (_Float16*)alloc((size_t)ND * 2);
  float* VB32 = (float*)alloc((size_t)ND * 4);
  _Float16* VBH = (_Float16*)alloc((size_t)ND * 2);
  float* ACC = (float*)alloc((size_t)ND * 4);

  const int GB_ND = (ND + 255) / 256;
  const int GB_N32 = N / 32;
  const int GB_ATB = N / ATB_NODES;

  // ---- projections, norms, z[n] = qn.ksum + N, C = kn^T qn ----------------
  k_zero<<<1, 256, 0, stream>>>(SC, 256);
  k_proj<<<GB_N32, 256, 0, stream>>>(preds, Wq, bq, Wk, bk, Q32, QH, KH, SC);
  k_finalize<<<1, 1, 0, stream>>>(SC);
  k_rowdot<<<N / 8, 256, 0, stream>>>(Q32, SC + SC_KSUM, SC + SC_INVFQK,
                                      (float)N, Z, N);
  k_zero<<<16, 256, 0, stream>>>(C_RAW, 4096);
  k_atb<<<GB_ATB, 256, 0, stream>>>(KH, QH, nullptr, C_RAW, nullptr);

  // ---- main hop-stack loop ------------------------------------------------
  // vs = preds ; out = alph[0]*preds
  k_fuse<<<GB_ND, 256, 0, stream>>>(preds, VA32, VAH, out, lin1 + 0, 1, ND);
  float* cur32 = VA32; _Float16* curh = VAH;
  float* alt32 = VB32; _Float16* alth = VBH;

  for (int i = 1; i < 4; ++i) {               // NITER-1
    k_fuse<<<GB_ND, 256, 0, stream>>>(cur32, nullptr, nullptr, ACC, lin2 + 0,
                                      1, ND);  // ACC = beta0*vs
    for (int j = 1; j < 4; ++j) {             // NITER_ATTN-1
      k_zero<<<16, 256, 0, stream>>>(KV, 4096);
      k_zero<<<1, 64, 0, stream>>>(SC + SC_CSUM, 64);
      k_atb<<<GB_ATB, 256, 0, stream>>>(KH, curh, cur32, KV, SC + SC_CSUM);
      k_stepB<<<16, 256, 0, stream>>>(C_RAW, KV, SC, B2H);
      k_bcast<<<GB_N32, 256, 0, stream>>>(QH, B2H, SC, Z, alt32, alth, ACC,
                                          lin2 + j);
      std::swap(cur32, alt32);
      std::swap(curh, alth);
    }
    k_zero<<<GB_ND, 256, 0, stream>>>(alt32, ND);
    k_spmm<<<(nnz * 16 + 255) / 256, 256, 0, stream>>>(rows, cols, vals, ACC,
                                                       alt32, nnz);
    k_fuse<<<GB_ND, 256, 0, stream>>>(alt32, nullptr, alth, out, lin1 + i, 0,
                                      ND);  // out += alph[i]*tmp
    std::swap(cur32, alt32);
    std::swap(curh, alth);
  }

  // ---- residual loss ------------------------------------------------------
  // gq accumulator (reuse ACC): GACC = beta[1]*query
  k_fuse<<<GB_ND, 256, 0, stream>>>(Q32, nullptr, nullptr, ACC, lin2 + 1, 1, ND);
  float* t32 = Q32; _Float16* th = QH;
  for (int k = 0; k < 2; ++k) {
    k_zero<<<16, 256, 0, stream>>>(KV, 4096);
    k_zero<<<1, 64, 0, stream>>>(SC + SC_CSUM, 64);
    k_atb<<<GB_ATB, 256, 0, stream>>>(KH, th, t32, KV, SC + SC_CSUM);
    k_stepB<<<16, 256, 0, stream>>>(C_RAW, KV, SC, B2H);
    k_bcast<<<GB_N32, 256, 0, stream>>>(QH, B2H, SC, Z, VB32, nullptr, nullptr,
                                        nullptr);  // o
    float* tgt32 = (k == 0) ? VA32 : Q32;
    _Float16* tgth = (k == 0) ? VAH : nullptr;
    k_zero<<<GB_ND, 256, 0, stream>>>(tgt32, ND);
    k_spmm<<<(nnz * 16 + 255) / 256, 256, 0, stream>>>(rows, cols, vals, VB32,
                                                       tgt32, nnz);
    k_fuse<<<GB_ND, 256, 0, stream>>>(tgt32, nullptr, tgth, ACC, lin2 + 2 + k,
                                      0, ND);  // GACC += beta[2+k]*t
    t32 = tgt32;
    th = tgth;
  }

  k_chunknorm<<<N / RCHUNK, 256, 0, stream>>>(ACC, SC + SC_SCH);
  k_rowdot<<<N / 8, 256, 0, stream>>>(ACC, SC + SC_KSUM, SC + SC_ONE, 0.f, Wn,
                                      N);  // w[n] = gq.ksum_raw
  k_zero<<<16, 256, 0, stream>>>(GRAW, 4096);
  k_atb<<<GB_ATB, 256, 0, stream>>>(KH, KH, nullptr, GRAW, nullptr);  // gram
  k_t1<<<256, 256, 0, stream>>>(ACC, GRAW, Wn, SC, N);
  k_t2t3<<<(nnz * 8 + 255) / 256, 256, 0, stream>>>(rows, cols, vals, ACC, KH,
                                                    Wn, SC, lin2, nnz);
  k_final<<<1, 1, 0, stream>>>(SC, lin2, out + (size_t)ND);
}
